// MultiHeadSelfAtten_18408229830994
// MI455X (gfx1250) — compile-verified, matrix operand order checked by reading, not run on hardware
//
#include <hip/hip_runtime.h>
#include <hip/hip_bf16.h>

#define B_    2
#define HEADS 16
#define HDIM  64
#define DIM   1024
#define SEQ   2048
#define PAST  512
#define TOT   2560

typedef __bf16 bf16_t;
typedef __attribute__((ext_vector_type(16))) __bf16 v16bf;
typedef __attribute__((ext_vector_type(8)))  float  v8f;

union FragAB { v16bf v; unsigned u[8]; };
union FragC  { v8f   v; float    f[8]; };

__device__ __forceinline__ bf16_t f2bf(float f) {
  union { float f; unsigned u; } x; x.f = f;
  unsigned r = x.u + 0x7FFFu + ((x.u >> 16) & 1u);      // round-to-nearest-even
  union { unsigned short s; bf16_t b; } y; y.s = (unsigned short)(r >> 16);
  return y.b;
}

// K-offset of the bf16 pair held in VGPR `v` for lane-half `half` (16-bit A/B
// fragment layout, CDNA5 ISA 7.12.2).
__device__ __forceinline__ int kbase(int v, int half) {
  return ((v & 4) << 2) + half * 8 + ((v & 3) << 1);
}

__device__ __forceinline__ v8f wmma_bf16(const FragAB& a, const FragAB& b, v8f c) {
  return __builtin_amdgcn_wmma_f32_16x16x32_bf16(false, a.v, false, b.v,
                                                 (short)0, c, false, false);
}

// CDNA5 async global->LDS copy (16B/lane, bypasses VGPRs, tracked by ASYNCcnt).
__device__ __forceinline__ void async_b128(const void* gptr, void* lptr) {
  unsigned loff = (unsigned)(size_t)lptr;   // low 32 bits of generic ptr = LDS offset
  asm volatile("global_load_async_to_lds_b128 %0, %1, off"
               :: "v"(loff), "v"(gptr) : "memory");
}
template <int N>
__device__ __forceinline__ void wait_async() {
  asm volatile("s_wait_asynccnt %0" :: "i"(N) : "memory");
}

// ---------------------------------------------------------------- elementwise
__global__ void cvt_f32_bf16(const float* __restrict__ src,
                             bf16_t* __restrict__ dst, int n) {
  int i = blockIdx.x * blockDim.x + threadIdx.x;
  if (i < n) dst[i] = f2bf(src[i]);
}

// past_keys/values [B,H,512,64] -> f32 outputs at t=0..511 + bf16 workspace
__global__ void copy_past(const float* __restrict__ pk, const float* __restrict__ pv,
                          float* __restrict__ kout, float* __restrict__ vout,
                          bf16_t* __restrict__ kbf, bf16_t* __restrict__ vbf) {
  int i = blockIdx.x * blockDim.x + threadIdx.x;      // < B*H*PAST*HDIM
  int d = i & 63, t = (i >> 6) & 511, h = (i >> 15) & 15, b = i >> 19;
  size_t dst = ((size_t)(b * HEADS + h) * TOT + t) * HDIM + d;
  float kv = pk[i], vv = pv[i];
  kout[dst] = kv; kbf[dst] = f2bf(kv);
  vout[dst] = vv; vbf[dst] = f2bf(vv);
}

// ------------------------------------------------------------------ QKV GEMM
// C[4096,3072] = x_bf16 @ Wqkv^T + bqkv, scattered into q/k/v.
// Double-buffered async global->LDS staging.
__global__ __launch_bounds__(256) void qkv_gemm(
    const bf16_t* __restrict__ xb, const bf16_t* __restrict__ wb,
    const float* __restrict__ bias,
    bf16_t* __restrict__ qbf,
    float* __restrict__ kout, bf16_t* __restrict__ kbf,
    float* __restrict__ vout, bf16_t* __restrict__ vbf) {
  __shared__ unsigned As[2][64][20];   // 64 rows x 32 bf16 (row padded to 80B)
  __shared__ unsigned Bs[2][64][20];
  const int tid = threadIdx.x, lane = tid & 31, w = tid >> 5;
  const int half = lane >> 4, mrow = lane & 15, nlane = lane & 15;
  const int wr = w >> 1, wc = w & 1;                  // 4x2 wave grid
  const int m0 = blockIdx.x * 64, n0 = blockIdx.y * 64;
  const int srow = tid >> 2, sc4 = tid & 3;           // 16B chunk per thread
  const char* xg = (const char*)xb;
  const char* wg = (const char*)wb;
  const size_t xrow = (size_t)(m0 + srow) * (DIM * 2) + sc4 * 16;
  const size_t wrow = (size_t)(n0 + srow) * (DIM * 2) + sc4 * 16;

  FragC acc0, acc1;
#pragma unroll
  for (int r = 0; r < 8; ++r) { acc0.f[r] = 0.f; acc1.f[r] = 0.f; }

  // prime panel 0
  async_b128(xg + xrow, &As[0][srow][sc4 * 4]);
  async_b128(wg + wrow, &Bs[0][srow][sc4 * 4]);

  for (int k0 = 0; k0 < DIM; k0 += 32) {
    const int cur = (k0 >> 5) & 1;
    if (k0 + 32 < DIM) {                // issue next panel, then wait for current
      async_b128(xg + xrow + (k0 + 32) * 2, &As[cur ^ 1][srow][sc4 * 4]);
      async_b128(wg + wrow + (k0 + 32) * 2, &Bs[cur ^ 1][srow][sc4 * 4]);
      __builtin_prefetch((const void*)(xg + xrow + (k0 + 64) * 2), 0, 0);
      wait_async<2>();                  // in-order: <=2 outstanding => cur landed
    } else {
      wait_async<0>();
    }
    __syncthreads();

    FragAB a, b0, b1;
#pragma unroll
    for (int v = 0; v < 8; ++v) {
      int kb = kbase(v, half) >> 1;
      a.u[v]  = As[cur][wr * 16 + mrow][kb];
      b0.u[v] = Bs[cur][wc * 32 + nlane][kb];
      b1.u[v] = Bs[cur][wc * 32 + 16 + nlane][kb];
    }
    acc0.v = wmma_bf16(a, b0, acc0.v);
    acc1.v = wmma_bf16(a, b1, acc1.v);
    __syncthreads();                    // reads done before buffer reuse
  }

#pragma unroll
  for (int cf = 0; cf < 2; ++cf) {
    FragC& acc = cf ? acc1 : acc0;
    int Ng = n0 + wc * 32 + cf * 16 + nlane;
    float bv = bias[Ng];
    int sec = Ng >> 10, j = Ng & 1023, hh = j >> 6, dd = j & 63;
#pragma unroll
    for (int r = 0; r < 8; ++r) {
      int Mg = m0 + wr * 16 + r + 8 * half;
      int bb = Mg >> 11, ss = Mg & 2047;
      float val = acc.f[r] + bv;
      if (sec == 0) {
        size_t qi = ((size_t)(bb * HEADS + hh) * SEQ + ss) * HDIM + dd;
        qbf[qi] = f2bf(val * 0.03125f);          // fold 1/sqrt(1024) into Q
      } else {
        size_t ki = ((size_t)(bb * HEADS + hh) * TOT + PAST + ss) * HDIM + dd;
        if (sec == 1) { kout[ki] = val; kbf[ki] = f2bf(val); }
        else          { vout[ki] = val; vbf[ki] = f2bf(val); }
      }
    }
  }
}

// ------------------------------------------------------------ flash attention
// 1 block = 4 waves = 64 queries of one (b,h); loop T in 32-key chunks.
__global__ __launch_bounds__(128) void flash_attn(
    const bf16_t* __restrict__ qbf, const bf16_t* __restrict__ kbf,
    const bf16_t* __restrict__ vbf, const int* __restrict__ mask,
    bf16_t* __restrict__ obf) {
  __shared__ unsigned Ks[32][36];      // K tile: [t 0..31][d as 32 dwords, 144B rows]
  __shared__ bf16_t   Vt[64 * 40];     // V^T tile: [d 0..63][t 0..31, pad]
  __shared__ bf16_t   Pb[4][16 * 40];  // per-wave P tile: [m 0..15][t 0..31, pad]
  const int tid = threadIdx.x, lane = tid & 31, w = tid >> 5;
  const int half = lane >> 4, mrow = lane & 15, nlane = lane & 15;
  const int b = blockIdx.x >> 4, h = blockIdx.x & 15;
  const int sbase = blockIdx.y * 64 + w * 16;
  const unsigned* qu = (const unsigned*)qbf;
  const unsigned* vu = (const unsigned*)vbf;
  const char*     kg = (const char*)kbf;

  FragAB qf[2];                                  // Q 16x64 as two 16x32 A-frags
#pragma unroll
  for (int dc = 0; dc < 2; ++dc)
#pragma unroll
    for (int v = 0; v < 8; ++v) {
      int d = dc * 32 + kbase(v, half);
      qf[dc].u[v] = qu[((size_t)(b * HEADS + h) * SEQ + (sbase + mrow)) * (HDIM / 2) + (d >> 1)];
    }

  FragC acc[4];
  float rowm[8], rowl[8];
#pragma unroll
  for (int r = 0; r < 8; ++r) { rowm[r] = -1e30f; rowl[r] = 0.f; }
#pragma unroll
  for (int dc = 0; dc < 4; ++dc)
#pragma unroll
    for (int r = 0; r < 8; ++r) acc[dc].f[r] = 0.f;

  const size_t kvb = (size_t)(b * HEADS + h) * TOT * (HDIM / 2);   // dwords
  for (int t0 = 0; t0 < TOT; t0 += 32) {
    __syncthreads();                             // prev reads done
    // K tile via async global->LDS (16B per lane, 2 issues cover 32x64 bf16)
#pragma unroll
    for (int j = 0; j < 2; ++j) {
      int q = tid + j * 128;
      int row = q >> 2, c4 = q & 3;
      async_b128(kg + (kvb + (size_t)(t0 + row) * 32) * 4 + c4 * 16,
                 &Ks[row][c4 * 4]);
    }
    // V tile staged transposed (manual: async copies cannot transpose)
#pragma unroll
    for (int i = 0; i < 8; ++i) {
      int q = tid + i * 128;
      int row = q >> 5, cd = q & 31;
      union { unsigned u; bf16_t b2[2]; } vv;
      vv.u = vu[kvb + (size_t)(t0 + row) * 32 + cd];
      Vt[(2 * cd) * 40 + row]     = vv.b2[0];
      Vt[(2 * cd + 1) * 40 + row] = vv.b2[1];
    }
    wait_async<0>();
    __syncthreads();

    FragC sc0, sc1;                              // scores for keys n / 16+n
#pragma unroll
    for (int r = 0; r < 8; ++r) { sc0.f[r] = 0.f; sc1.f[r] = 0.f; }
#pragma unroll
    for (int dc = 0; dc < 2; ++dc) {
      FragAB bk0, bk1;
#pragma unroll
      for (int v = 0; v < 8; ++v) {
        int kb = kbase(v, half) >> 1;
        bk0.u[v] = Ks[nlane][dc * 16 + kb];
        bk1.u[v] = Ks[16 + nlane][dc * 16 + kb];
      }
      sc0.v = wmma_bf16(qf[dc], bk0, sc0.v);
      sc1.v = wmma_bf16(qf[dc], bk1, sc1.v);
    }

#pragma unroll
    for (int r = 0; r < 8; ++r) {                // mask + online softmax
      int sg = sbase + r + 8 * half;
      const int* mp = mask + ((size_t)b * SEQ + sg) * TOT + t0;
      if (mp[nlane] == 0)      sc0.f[r] = -1000.f;
      if (mp[16 + nlane] == 0) sc1.f[r] = -1000.f;
      float mx = fmaxf(sc0.f[r], sc1.f[r]);
      mx = fmaxf(mx, __shfl_xor(mx, 1, 32));
      mx = fmaxf(mx, __shfl_xor(mx, 2, 32));
      mx = fmaxf(mx, __shfl_xor(mx, 4, 32));
      mx = fmaxf(mx, __shfl_xor(mx, 8, 32));
      float mnew = fmaxf(rowm[r], mx);
      float corr = __expf(rowm[r] - mnew);
      rowm[r] = mnew;
      float p0 = __expf(sc0.f[r] - mnew);
      float p1 = __expf(sc1.f[r] - mnew);
      float rs = p0 + p1;
      rs += __shfl_xor(rs, 1, 32);
      rs += __shfl_xor(rs, 2, 32);
      rs += __shfl_xor(rs, 4, 32);
      rs += __shfl_xor(rs, 8, 32);
      rowl[r] = rowl[r] * corr + rs;
#pragma unroll
      for (int dc = 0; dc < 4; ++dc) acc[dc].f[r] *= corr;
      Pb[w][(r + 8 * half) * 40 + nlane]      = f2bf(p0);   // C-layout -> LDS
      Pb[w][(r + 8 * half) * 40 + 16 + nlane] = f2bf(p1);
    }
    asm volatile("s_wait_dscnt 0" ::: "memory"); // wave-private P RAW in LDS

    FragAB pf;
#pragma unroll
    for (int v = 0; v < 8; ++v)                  // reload P as A-fragment
      pf.u[v] = *(const unsigned*)&Pb[w][mrow * 40 + kbase(v, half)];
#pragma unroll
    for (int dc = 0; dc < 4; ++dc) {             // O += P @ V
      FragAB bv;
#pragma unroll
      for (int v = 0; v < 8; ++v)
        bv.u[v] = *(const unsigned*)&Vt[(dc * 16 + nlane) * 40 + kbase(v, half)];
      acc[dc].v = wmma_bf16(pf, bv, acc[dc].v);
    }
  }

#pragma unroll
  for (int dc = 0; dc < 4; ++dc)                 // normalize + merge heads
#pragma unroll
    for (int r = 0; r < 8; ++r) {
      int sg = sbase + r + 8 * half;
      int d = h * HDIM + dc * 16 + nlane;
      obf[(size_t)(b * SEQ + sg) * DIM + d] = f2bf(acc[dc].f[r] / rowl[r]);
    }
}

// ------------------------------------------------------------- output GEMM
__global__ __launch_bounds__(256) void out_gemm(
    const bf16_t* __restrict__ ab, const bf16_t* __restrict__ wb,
    const float* __restrict__ bias, float* __restrict__ out) {
  __shared__ unsigned As[2][64][20];
  __shared__ unsigned Bs[2][64][20];
  const int tid = threadIdx.x, lane = tid & 31, w = tid >> 5;
  const int half = lane >> 4, mrow = lane & 15, nlane = lane & 15;
  const int wr = w >> 1, wc = w & 1;
  const int m0 = blockIdx.x * 64, n0 = blockIdx.y * 64;
  const int srow = tid >> 2, sc4 = tid & 3;
  const char* ag = (const char*)ab;
  const char* wg = (const char*)wb;
  const size_t arow = (size_t)(m0 + srow) * (DIM * 2) + sc4 * 16;
  const size_t wrow = (size_t)(n0 + srow) * (DIM * 2) + sc4 * 16;

  FragC acc0, acc1;
#pragma unroll
  for (int r = 0; r < 8; ++r) { acc0.f[r] = 0.f; acc1.f[r] = 0.f; }

  async_b128(ag + arow, &As[0][srow][sc4 * 4]);
  async_b128(wg + wrow, &Bs[0][srow][sc4 * 4]);

  for (int k0 = 0; k0 < DIM; k0 += 32) {
    const int cur = (k0 >> 5) & 1;
    if (k0 + 32 < DIM) {
      async_b128(ag + arow + (k0 + 32) * 2, &As[cur ^ 1][srow][sc4 * 4]);
      async_b128(wg + wrow + (k0 + 32) * 2, &Bs[cur ^ 1][srow][sc4 * 4]);
      __builtin_prefetch((const void*)(ag + arow + (k0 + 64) * 2), 0, 0);
      wait_async<2>();
    } else {
      wait_async<0>();
    }
    __syncthreads();

    FragAB a, b0, b1;
#pragma unroll
    for (int v = 0; v < 8; ++v) {
      int kb = kbase(v, half) >> 1;
      a.u[v]  = As[cur][wr * 16 + mrow][kb];
      b0.u[v] = Bs[cur][wc * 32 + nlane][kb];
      b1.u[v] = Bs[cur][wc * 32 + 16 + nlane][kb];
    }
    acc0.v = wmma_bf16(a, b0, acc0.v);
    acc1.v = wmma_bf16(a, b1, acc1.v);
    __syncthreads();
  }

#pragma unroll
  for (int cf = 0; cf < 2; ++cf) {
    FragC& acc = cf ? acc1 : acc0;
    int Ng = n0 + wc * 32 + cf * 16 + nlane;
    float bv = bias[Ng];
#pragma unroll
    for (int r = 0; r < 8; ++r) {
      int Mg = m0 + wr * 16 + r + 8 * half;
      out[(size_t)Mg * DIM + Ng] = acc.f[r] + bv;
    }
  }
}

// ------------------------------------------------------------------- launch
extern "C" void kernel_launch(void* const* d_in, const int* in_sizes, int n_in,
                              void* d_out, int out_size, void* d_ws, size_t ws_size,
                              hipStream_t stream) {
  const float* x    = (const float*)d_in[0];
  const int*   mask = (const int*)  d_in[1];
  const float* pk   = (const float*)d_in[2];
  const float* pv   = (const float*)d_in[3];
  const float* Wqkv = (const float*)d_in[4];
  const float* bqkv = (const float*)d_in[5];
  const float* Wout = (const float*)d_in[6];
  const float* bout = (const float*)d_in[7];

  float* out_f = (float*)d_out;                                   // [B,S,DIM]
  float* k_out = out_f + (size_t)B_ * SEQ * DIM;                  // [B,H,T,Dh]
  float* v_out = k_out + (size_t)B_ * HEADS * TOT * HDIM;

  char* ws = (char*)d_ws;
  bf16_t* x_bf    = (bf16_t*)(ws);                  //  8 MB  x       bf16
  bf16_t* wqkv_bf = (bf16_t*)(ws + (8ull  << 20));  //  6 MB  Wqkv    bf16
  bf16_t* wout_bf = (bf16_t*)(ws + (14ull << 20));  //  2 MB  Wout    bf16
  bf16_t* q_bf    = (bf16_t*)(ws + (16ull << 20));  //  8 MB  q*scale bf16
  bf16_t* k_bf    = (bf16_t*)(ws + (24ull << 20));  // 10 MB  k       bf16
  bf16_t* v_bf    = (bf16_t*)(ws + (34ull << 20));  // 10 MB  v       bf16
  bf16_t* o_bf    = (bf16_t*)(ws + (44ull << 20));  //  8 MB  attn out bf16

  int nx = B_ * SEQ * DIM;
  cvt_f32_bf16<<<(nx + 255) / 256, 256, 0, stream>>>(x, x_bf, nx);
  int nwq = 3 * DIM * DIM;
  cvt_f32_bf16<<<(nwq + 255) / 256, 256, 0, stream>>>(Wqkv, wqkv_bf, nwq);
  int nwo = DIM * DIM;
  cvt_f32_bf16<<<(nwo + 255) / 256, 256, 0, stream>>>(Wout, wout_bf, nwo);

  int npast = B_ * HEADS * PAST * HDIM;
  copy_past<<<npast / 256, 256, 0, stream>>>(pk, pv, k_out, v_out, k_bf, v_bf);

  qkv_gemm<<<dim3(B_ * SEQ / 64, 3 * DIM / 64), 256, 0, stream>>>(
      x_bf, wqkv_bf, bqkv, q_bf, k_out, k_bf, v_out, v_bf);

  flash_attn<<<dim3(B_ * HEADS, SEQ / 64), 128, 0, stream>>>(
      q_bf, k_bf, v_bf, mask, o_bf);

  out_gemm<<<dim3(B_ * SEQ / 64, DIM / 64), 256, 0, stream>>>(
      o_bf, wout_bf, bout, out_f);
}
